// MultiHeadAttention_38938173506175
// MI455X (gfx1250) — matrix operand order repaired and checked by reading, not compiled, not run on hardware
//
#include <hip/hip_runtime.h>
#include <hip/hip_bf16.h>

// MI455X / gfx1250, wave32. All matmuls through v_wmma_f32_16x16x32_bf16.
// N=2048, D_IN=1024, H=16, E=64.

typedef __bf16 bf16;
typedef __attribute__((ext_vector_type(8)))  __bf16 bf16x8;
typedef __attribute__((ext_vector_type(16))) __bf16 bf16x16;
typedef __attribute__((ext_vector_type(8)))  float  f32x8;

#define N_TOK 2048
#define D_IN  1024
#define H     16
#define E     64

static __device__ __forceinline__ bf16x16 load_a_frag(const bf16* p_lo, const bf16* p_hi) {
  // A 16x32 bf16 frag as used here: lane m (both halves) holds K = h*16+0..15 (contiguous, as B)
  bf16x16 r;
  reinterpret_cast<bf16x8*>(&r)[0] = *reinterpret_cast<const bf16x8*>(p_lo);
  reinterpret_cast<bf16x8*>(&r)[1] = *reinterpret_cast<const bf16x8*>(p_hi);
  return r;
}

static __device__ __forceinline__ bf16x16 load_b_frag(const bf16* p) {
  // B 32x16 bf16 layout: lane n holds 16 contiguous K values (half selects K 0-15 / 16-31)
  return *reinterpret_cast<const bf16x16*>(p);
}

static __device__ __forceinline__ f32x8 wmma_bf16(bf16x16 a, bf16x16 b, f32x8 c) {
  return __builtin_amdgcn_wmma_f32_16x16x32_bf16(false, a, false, b, (short)0, c, false, false);
}

// ---------------- conversion / transpose kernels ----------------

__global__ void k_convert_x(const float* __restrict__ x, bf16* __restrict__ xb, int n) {
  int i = blockIdx.x * blockDim.x + threadIdx.x;
  if (i < n) xb[i] = (bf16)x[i];
}

// q/k/v_proj: (H, D_IN, E) -> wt[c][d], c = h*64+e  (N x K row-major, i.e. B transposed)
__global__ void k_transpose_qkv(const float* __restrict__ w, bf16* __restrict__ wt) {
  int t = blockIdx.x * blockDim.x + threadIdx.x;   // 1024*1024
  int d = t & 1023;
  int c = t >> 10;
  wt[t] = (bf16)w[((c >> 6) << 16) + d * 64 + (c & 63)];
}

// o_proj: (D_IN, D_IN) row-major [d][c] -> ot[c][d]
__global__ void k_transpose_o(const float* __restrict__ o, bf16* __restrict__ ot) {
  int t = blockIdx.x * blockDim.x + threadIdx.x;
  int d = t & 1023;
  int c = t >> 10;
  ot[t] = (bf16)o[d * 1024 + c];
}

// ---------------- generic bf16 WMMA GEMM ----------------
// C(MxN) = A(MxK, row-major bf16) * Bt(NxK, row-major bf16)^T
// Per-wave tile: 32x64 (2 m-subtiles x 4 n-subtiles), software-pipelined in registers.
// store_mode: 0 = bf16 row-major (ldc), 1 = bf16 transposed C[c*ldc+m], 2 = f32 row-major
__global__ __launch_bounds__(32) void k_gemm(const bf16* __restrict__ A,
                                             const bf16* __restrict__ Bt,
                                             void* __restrict__ C,
                                             int M, int N, int K,
                                             int store_mode, int ldc) {
  const int lane = threadIdx.x;
  const int half = lane >> 4;
  const int li   = lane & 15;
  const int mt   = blockIdx.x * 32;
  const int nt   = blockIdx.y * 64;

  const bf16* a0 = A + (size_t)(mt + li) * K;
  const bf16* a1 = A + (size_t)(mt + 16 + li) * K;

  f32x8 acc[2][4] = {};
  bf16x16 aF[2][2], bF[2][4];

  auto load_frags = [&](int buf, int kc) {
    aF[buf][0] = load_a_frag(a0 + kc + half * 16, a0 + kc + half * 16 + 8);
    aF[buf][1] = load_a_frag(a1 + kc + half * 16, a1 + kc + half * 16 + 8);
#pragma unroll
    for (int j = 0; j < 4; ++j)
      bF[buf][j] = load_b_frag(Bt + (size_t)(nt + j * 16 + li) * K + kc + half * 16);
  };
  auto compute = [&](int buf) {
#pragma unroll
    for (int i = 0; i < 2; ++i)
#pragma unroll
      for (int j = 0; j < 4; ++j)
        acc[i][j] = wmma_bf16(aF[buf][i], bF[buf][j], acc[i][j]);
  };

  load_frags(0, 0);
  int kc = 0;
  for (; kc + 64 < K; kc += 64) {
    load_frags(1, kc + 32);                         // loads for next half-step in flight
    __builtin_prefetch(a0 + kc + 256, 0, 3);        // near-scope global_prefetch_b8
    compute(0);                                     // 8 wmmas overlap those loads
    load_frags(0, kc + 64);
    compute(1);
  }
  load_frags(1, kc + 32);
  compute(0);
  compute(1);

  // C layout: lane li = column n, VGPR r = row (16*i + half*8 + r)
  if (store_mode == 2) {
    float* Cf = (float*)C;
#pragma unroll
    for (int i = 0; i < 2; ++i)
#pragma unroll
      for (int j = 0; j < 4; ++j)
#pragma unroll
        for (int r = 0; r < 8; ++r)
          Cf[(size_t)(mt + i * 16 + half * 8 + r) * ldc + nt + j * 16 + li] = acc[i][j][r];
  } else if (store_mode == 0) {
    bf16* Cb = (bf16*)C;
#pragma unroll
    for (int i = 0; i < 2; ++i)
#pragma unroll
      for (int j = 0; j < 4; ++j)
#pragma unroll
        for (int r = 0; r < 8; ++r)
          Cb[(size_t)(mt + i * 16 + half * 8 + r) * ldc + nt + j * 16 + li] = (bf16)acc[i][j][r];
  } else {  // transposed: contiguous 16B store per lane per subtile
    bf16* Cb = (bf16*)C;
#pragma unroll
    for (int i = 0; i < 2; ++i)
#pragma unroll
      for (int j = 0; j < 4; ++j) {
        bf16x8 v;
#pragma unroll
        for (int r = 0; r < 8; ++r) v[r] = (bf16)acc[i][j][r];
        *reinterpret_cast<bf16x8*>(Cb + (size_t)(nt + j * 16 + li) * ldc +
                                   mt + i * 16 + half * 8) = v;
      }
  }
}

// ---------------- flash attention ----------------
// One wave per (head, 16-query tile). Qb/Kb: (n, h*64+e) bf16. Vt: [h*64+e][n] bf16.
__global__ __launch_bounds__(32) void k_attn(const bf16* __restrict__ Qb,
                                             const bf16* __restrict__ Kb,
                                             const bf16* __restrict__ Vt,
                                             bf16* __restrict__ ctxb) {
  __shared__ __align__(64) bf16 pbuf[16 * 32];   // P transpose bounce (C-layout -> A-layout)

  const int lane = threadIdx.x;
  const int half = lane >> 4;
  const int li   = lane & 15;
  const int qt   = blockIdx.x;   // 0..127
  const int hd   = blockIdx.y;   // 0..15

  // Q A-fragments for feature chunks 0..31 and 32..63 (loaded once)
  const bf16* qrow = Qb + (size_t)(qt * 16 + li) * D_IN + hd * E;
  bf16x16 aq0 = load_a_frag(qrow + half * 16,      qrow + half * 16 + 8);
  bf16x16 aq1 = load_a_frag(qrow + 32 + half * 16, qrow + 40 + half * 16);

  f32x8 ctx[4] = {};
  float mrun[8], lrun[8];
#pragma unroll
  for (int r = 0; r < 8; ++r) { mrun[r] = -1e30f; lrun[r] = 0.f; }

  for (int ch = 0; ch < N_TOK; ch += 32) {
    // prefetch next chunk's K rows and V rows (L2-resident, keep near)
    if (ch + 32 < N_TOK) {
      __builtin_prefetch(Kb + (size_t)(ch + 32 + li) * D_IN + hd * E, 0, 3);
      __builtin_prefetch(Vt + (size_t)(hd * E + li) * N_TOK + ch + 32, 0, 3);
    }
    // ---- scores: two 16-key tiles, K=64 via two chained wmmas each ----
    f32x8 s[2];
#pragma unroll
    for (int t = 0; t < 2; ++t) {
      const bf16* krow = Kb + (size_t)(ch + t * 16 + li) * D_IN + hd * E;
      f32x8 z = {};
      s[t] = wmma_bf16(aq0, load_b_frag(krow + half * 16), z);
      s[t] = wmma_bf16(aq1, load_b_frag(krow + 32 + half * 16), s[t]);
    }
    // scale by 1/sqrt(64)
#pragma unroll
    for (int t = 0; t < 2; ++t)
#pragma unroll
      for (int r = 0; r < 8; ++r) s[t][r] *= 0.125f;

    // ---- online softmax: row stats live per half (rows half*8+r across 16 lanes) ----
    float mnew[8], alpha[8];
#pragma unroll
    for (int r = 0; r < 8; ++r) {
      float v = fmaxf(s[0][r], s[1][r]);
      v = fmaxf(v, __shfl_xor(v, 1, 16));
      v = fmaxf(v, __shfl_xor(v, 2, 16));
      v = fmaxf(v, __shfl_xor(v, 4, 16));
      v = fmaxf(v, __shfl_xor(v, 8, 16));
      mnew[r]  = fmaxf(mrun[r], v);
      alpha[r] = __expf(mrun[r] - mnew[r]);
      mrun[r]  = mnew[r];
    }
#pragma unroll
    for (int r = 0; r < 8; ++r) {
      s[0][r] = __expf(s[0][r] - mnew[r]);
      s[1][r] = __expf(s[1][r] - mnew[r]);
      float rs = s[0][r] + s[1][r];
      rs += __shfl_xor(rs, 1, 16);
      rs += __shfl_xor(rs, 2, 16);
      rs += __shfl_xor(rs, 4, 16);
      rs += __shfl_xor(rs, 8, 16);
      lrun[r] = lrun[r] * alpha[r] + rs;
    }
    // rescale running context
#pragma unroll
    for (int j = 0; j < 4; ++j)
#pragma unroll
      for (int r = 0; r < 8; ++r) ctx[j][r] *= alpha[r];

    // ---- transpose P (16x32) from C-layout to A-layout through LDS ----
    __syncthreads();
#pragma unroll
    for (int t = 0; t < 2; ++t)
#pragma unroll
      for (int r = 0; r < 8; ++r)
        pbuf[(half * 8 + r) * 32 + t * 16 + li] = (bf16)s[t][r];
    __syncthreads();
    const bf16* prow = pbuf + li * 32;
    bf16x16 ap = load_a_frag(prow + half * 16, prow + half * 16 + 8);

    // ---- ctx += P * V  (B from transposed V: contiguous per-lane loads) ----
#pragma unroll
    for (int j = 0; j < 4; ++j) {
      const bf16* vrow = Vt + (size_t)(hd * E + j * 16 + li) * N_TOK + ch + half * 16;
      ctx[j] = wmma_bf16(ap, load_b_frag(vrow), ctx[j]);
    }
  }

  // ---- normalize and store concat-layout bf16 context ----
#pragma unroll
  for (int r = 0; r < 8; ++r) {
    float inv = 1.0f / lrun[r];
#pragma unroll
    for (int j = 0; j < 4; ++j)
      ctxb[(size_t)(qt * 16 + half * 8 + r) * D_IN + hd * E + j * 16 + li] =
          (bf16)(ctx[j][r] * inv);
  }
}

// ---------------- host launch ----------------

extern "C" void kernel_launch(void* const* d_in, const int* in_sizes, int n_in,
                              void* d_out, int out_size, void* d_ws, size_t ws_size,
                              hipStream_t stream) {
  const float* x  = (const float*)d_in[0];
  const float* qp = (const float*)d_in[1];
  const float* kp = (const float*)d_in[2];
  const float* vp = (const float*)d_in[3];
  const float* op = (const float*)d_in[4];
  float* out = (float*)d_out;

  // workspace layout (28 MiB total)
  char* ws = (char*)d_ws;
  bf16* xb   = (bf16*)(ws + (size_t)0);            // 4 MiB : x bf16 (2048x1024)
  bf16* wqt  = (bf16*)(ws + ((size_t)4  << 20));   // 2 MiB : Wq^T (1024x1024)
  bf16* wkt  = (bf16*)(ws + ((size_t)6  << 20));   // 2 MiB
  bf16* wvt  = (bf16*)(ws + ((size_t)8  << 20));   // 2 MiB
  bf16* ot   = (bf16*)(ws + ((size_t)10 << 20));   // 2 MiB : o_proj^T
  bf16* Qb   = (bf16*)(ws + ((size_t)12 << 20));   // 4 MiB : Q (n, h*64+e)
  bf16* Kb   = (bf16*)(ws + ((size_t)16 << 20));   // 4 MiB : K (n, h*64+e)
  bf16* Vt   = (bf16*)(ws + ((size_t)20 << 20));   // 4 MiB : V^T [h*64+e][n]
  bf16* ctxb = (bf16*)(ws + ((size_t)24 << 20));   // 4 MiB : concat ctx (n, h*64+e)

  // conversions
  k_convert_x<<<(N_TOK * D_IN) / 256, 256, 0, stream>>>(x, xb, N_TOK * D_IN);
  k_transpose_qkv<<<(D_IN * D_IN) / 256, 256, 0, stream>>>(qp, wqt);
  k_transpose_qkv<<<(D_IN * D_IN) / 256, 256, 0, stream>>>(kp, wkt);
  k_transpose_qkv<<<(D_IN * D_IN) / 256, 256, 0, stream>>>(vp, wvt);
  k_transpose_o  <<<(D_IN * D_IN) / 256, 256, 0, stream>>>(op, ot);

  // QKV projections (2048x1024x1024 each), 32x64 tile per wave
  dim3 gg(N_TOK / 32, D_IN / 64);
  k_gemm<<<gg, 32, 0, stream>>>(xb, wqt, Qb, N_TOK, D_IN, D_IN, /*bf16 rowmajor*/0, D_IN);
  k_gemm<<<gg, 32, 0, stream>>>(xb, wkt, Kb, N_TOK, D_IN, D_IN, 0, D_IN);
  k_gemm<<<gg, 32, 0, stream>>>(xb, wvt, Vt, N_TOK, D_IN, D_IN, /*transposed*/1, N_TOK);

  // flash attention: (128 query tiles) x (16 heads), one wave each
  k_attn<<<dim3(N_TOK / 16, H), 32, 0, stream>>>(Qb, Kb, Vt, ctxb);

  // output projection -> fp32
  k_gemm<<<gg, 32, 0, stream>>>(ctxb, ot, out, N_TOK, D_IN, D_IN, /*f32*/2, D_IN);
}
